// OptimizedTritonAdaptivePiecewiseConv2d_88519275970722
// MI455X (gfx1250) — compile-verified, matrix operand order checked by reading, not run on hardware
//
#include <hip/hip_runtime.h>

// ---------------------------------------------------------------------------
// Piecewise-linear "adaptive" conv as implicit GEMM on the CDNA5 matrix pipe
// (v_wmma_f32_16x16x32_f16, wave32).
//
//   out[b,oc,s] = bias[oc] + sum_k s0[oc,k]*g0(x) + sum_k s1[oc,k]*g1(x)
//   g0(x) = clip(x,p0,p1)-p0 ,  g1(x) = clip(x,p1,p2)-p1   (positions shared)
//
// GEMM shape: M=64 (OC), N=73728 (B*OH*OW), K=576 (2*CIN*KH*KW).
// Round-3 change: 128-thread WGs + fully-unrolled K loop so every im2col
// index is a compile-time constant (no per-element division chains), clamps
// via v_med3_f32, validity via 9 precomputed per-thread booleans.
// ---------------------------------------------------------------------------

typedef __attribute__((ext_vector_type(16))) _Float16 v16h;
typedef __attribute__((ext_vector_type(8)))  _Float16 v8h;
typedef __attribute__((ext_vector_type(8)))  float    v8f;

#define OC_    64
#define CIN_   32
#define H_     96
#define W_     96
#define B_     8
#define KCONV  288              // CIN*KH*KW
#define KK2    576              // 2*KCONV
#define S_     9216             // OH*OW (96*96)
#define NTOT   (B_ * S_)        // 73728
#define TILE_N 128
#define KT_ABS 18               // 576 / 32 (absolute K-steps, for A layout)
#define KT_OUT 9                // outer steps: each covers g0 kt and g1 kt+9
#define BT_STRIDE 72            // f16/row: 64 K-values + 8 pad (144B, 16B-align)

// ---------------------------------------------------------------------------
// (oc, kk) -> flat index into the WMMA-A-fragment-ordered buffer.
// A fragment (16x32 f16, wave32): lane L holds M = L%16, half = L/16;
// element j: K = (j/8)*16 + half*8 + (j%8)   (two contiguous 8-K runs).
// Storage: Aws[((mt*18 + kt)*32 + L)*16 + j], one 32B blob per lane.
// ---------------------------------------------------------------------------
__device__ __forceinline__ int frag_index(int oc, int kk) {
    int mt      = oc >> 4;
    int lane_lo = oc & 15;
    int kt      = kk >> 5;
    int kw32    = kk & 31;
    int group   = kw32 >> 4;
    int r       = kw32 & 15;
    int half    = r >> 3;
    int off     = r & 7;
    int L       = half * 16 + lane_lo;
    int j       = group * 8 + off;
    return ((mt * KT_ABS + kt) * 32 + L) * 16 + j;
}

// ---------------------------------------------------------------------------
// Prepass: slopes (reference semantics: dp>0 ? dv/dp : 0) -> f16 in fragment
// order. g0 slopes at kk=k, g1 slopes at kk=288+k.
// ---------------------------------------------------------------------------
__global__ void pack_slopes_kernel(const float* __restrict__ pos,
                                   const float* __restrict__ val,
                                   _Float16* __restrict__ Aws) {
    int tid = blockIdx.x * 256 + threadIdx.x;
    if (tid >= OC_ * KCONV) return;
    int oc = tid / KCONV;
    int k  = tid - oc * KCONV;
    int base = tid * 3;                         // (oc*288 + k) * 3
    float p0 = pos[base + 0], p1 = pos[base + 1], p2 = pos[base + 2];
    float v0 = val[base + 0], v1 = val[base + 1], v2 = val[base + 2];
    float dp0 = p1 - p0, dp1 = p2 - p1;
    float s0 = (dp0 > 0.0f) ? (v1 - v0) / dp0 : 0.0f;
    float s1 = (dp1 > 0.0f) ? (v2 - v1) / dp1 : 0.0f;
    Aws[frag_index(oc, k)]         = (_Float16)s0;
    Aws[frag_index(oc, KCONV + k)] = (_Float16)s1;
}

// ---------------------------------------------------------------------------
// bias[oc] = sum_k values[oc,k,0]   (one wave32 per oc, shuffle reduce)
// ---------------------------------------------------------------------------
__global__ void bias_kernel(const float* __restrict__ val,
                            float* __restrict__ bias) {
    int oc   = blockIdx.x;
    int lane = threadIdx.x;
    float s = 0.0f;
    for (int k = lane; k < KCONV; k += 32)
        s += val[(oc * KCONV + k) * 3];
    #pragma unroll
    for (int off = 16; off > 0; off >>= 1)
        s += __shfl_down(s, off, 32);
    if (lane == 0) bias[oc] = s;
}

// ---------------------------------------------------------------------------
// Main kernel: one WG (128 thr = 4 waves) computes a 64(OC) x 128(N) tile.
// Wave w: M-tile mt = w, N-tiles 0..7 -> 8 accumulators (high WMMA ILP).
// Builder: thread owns one n column, builds all 32 k of the step, with the
// outer 9-step loop fully unrolled so k (-> cin,kh,kw, x-offset, validity
// selector) is a compile-time constant.
// ---------------------------------------------------------------------------
__global__ __launch_bounds__(128)
void pwconv_wmma_kernel(const float* __restrict__ x,
                        const float* __restrict__ pos,
                        const _Float16* __restrict__ Aws,
                        const float* __restrict__ bias,
                        float* __restrict__ out) {
    __shared__ __align__(32) _Float16 Bt[TILE_N * BT_STRIDE];   // 18,432 B

    const int tid    = threadIdx.x;
    const int n_base = blockIdx.x * TILE_N;       // global n = b*S + s
    const int b      = n_base / S_;               // tile never crosses batch
    const int s_base = n_base - b * S_;

    const float p0 = pos[0], p1 = pos[1], p2 = pos[2];   // shared positions

    const int mt = tid >> 5;        // wave id == M tile (16 oc rows)
    const int L  = tid & 31;        // lane

    // Builder role: this thread owns n column `tid`, all k of each step.
    const int s_n = s_base + tid;
    const int oh  = s_n / W_;
    const int ow  = s_n - oh * W_;
    const float* xb = x + (size_t)b * (CIN_ * H_ * W_);
    const int tbase = (oh - 1) * W_ + (ow - 1);   // add cin*9216+kh*96+kw (const)

    // 9 validity booleans, one per (kh,kw) combo; kh/kw==1 is always valid.
    const bool vh0 = (oh >= 1), vh2 = (oh <= H_ - 2);
    const bool vw0 = (ow >= 1), vw2 = (ow <= W_ - 2);
    const bool vmask[9] = { vh0 && vw0, vh0, vh0 && vw2,
                            vw0,        true, vw2,
                            vh2 && vw0, vh2, vh2 && vw2 };

    v8f acc[8] = {};

    #pragma unroll
    for (int ktO = 0; ktO < KT_OUT; ++ktO) {
        // A fragments first: global latency overlaps the builder below.
        v16h afrag0 = *(const v16h*)(Aws + (((mt * KT_ABS + ktO) * 32 + L) << 4));
        v16h afrag1 = *(const v16h*)(Aws + (((mt * KT_ABS + KT_OUT + ktO) * 32 + L) << 4));

        // ---- builder: 32 k values, k compile-time; one x load feeds g0+g1 ----
        {
            v8h g0v[4] = {}, g1v[4] = {};
            #pragma unroll
            for (int q = 0; q < 32; ++q) {
                const int k   = ktO * 32 + q;        // compile-time
                const int cin = k / 9;
                const int rem = k - cin * 9;         // (kh,kw) selector
                const int kh  = rem / 3;
                const int kw  = rem - kh * 3;
                const int koff = cin * (H_ * W_) + kh * W_ + kw;  // immediate
                const bool valid = vmask[rem];       // constant-indexed
                int addr = tbase + koff;
                addr = valid ? addr : 0;             // cndmask: safe address
                float v = xb[addr];
                v = valid ? v : 0.0f;                // cndmask: zero-pad
                float a0 = __builtin_amdgcn_fmed3f(v, p0, p1) - p0;  // clamp
                float a1 = __builtin_amdgcn_fmed3f(v, p1, p2) - p1;
                g0v[q >> 3][q & 7] = (_Float16)a0;
                g1v[q >> 3][q & 7] = (_Float16)a1;
            }
            _Float16* dst = &Bt[tid * BT_STRIDE];
            *(v8h*)(dst)      = g0v[0];   // g0: elems 0..31
            *(v8h*)(dst + 8)  = g0v[1];
            *(v8h*)(dst + 16) = g0v[2];
            *(v8h*)(dst + 24) = g0v[3];
            *(v8h*)(dst + 32) = g1v[0];   // g1: elems 32..63
            *(v8h*)(dst + 40) = g1v[1];
            *(v8h*)(dst + 48) = g1v[2];
            *(v8h*)(dst + 56) = g1v[3];
        }
        __syncthreads();

        // ---- 2 K-steps x 8 N-tiles = 16 WMMAs; A reused 8x each ----
        #pragma unroll
        for (int t = 0; t < 8; ++t) {
            const _Float16* brow =
                &Bt[(t * 16 + (L & 15)) * BT_STRIDE + (L >> 4) * 8];
            v8h blo = *(const v8h*)(brow);          // g0 runs: +0, +16
            v8h bhi = *(const v8h*)(brow + 16);
            v16h bf0 = __builtin_shufflevector(
                blo, bhi, 0, 1, 2, 3, 4, 5, 6, 7, 8, 9, 10, 11, 12, 13, 14, 15);
            acc[t] = __builtin_amdgcn_wmma_f32_16x16x32_f16(
                         false, afrag0, false, bf0, (short)0, acc[t], false, false);
            v8h clo = *(const v8h*)(brow + 32);     // g1 runs: +32, +48
            v8h chi = *(const v8h*)(brow + 48);
            v16h bf1 = __builtin_shufflevector(
                clo, chi, 0, 1, 2, 3, 4, 5, 6, 7, 8, 9, 10, 11, 12, 13, 14, 15);
            acc[t] = __builtin_amdgcn_wmma_f32_16x16x32_f16(
                         false, afrag1, false, bf1, (short)0, acc[t], false, false);
        }
        __syncthreads();
    }

    // ---- epilogue: C/D layout  M = v + 8*(L/16), N = L%16 ----
    const int half = L >> 4;
    float bl[8];
    #pragma unroll
    for (int v = 0; v < 8; ++v)
        bl[v] = bias[mt * 16 + half * 8 + v];
    #pragma unroll
    for (int t = 0; t < 8; ++t) {
        int s_out = s_base + t * 16 + (L & 15);
        #pragma unroll
        for (int v = 0; v < 8; ++v) {
            int oc = mt * 16 + half * 8 + v;
            out[((size_t)b * OC_ + oc) * S_ + s_out] = acc[t][v] + bl[v];
        }
    }
}

// ---------------------------------------------------------------------------
extern "C" void kernel_launch(void* const* d_in, const int* in_sizes, int n_in,
                              void* d_out, int out_size, void* d_ws, size_t ws_size,
                              hipStream_t stream) {
    const float* x   = (const float*)d_in[0];   // [8,32,96,96]
    const float* pos = (const float*)d_in[1];   // [64,32,3,3,3]
    const float* val = (const float*)d_in[2];   // [64,32,3,3,3]
    float* out = (float*)d_out;                 // [8,64,96,96]

    _Float16* Aws = (_Float16*)d_ws;                               // 73,728 B
    float* bias   = (float*)((char*)d_ws + (size_t)OC_ * KK2 * 2); // 256 B

    pack_slopes_kernel<<<(OC_ * KCONV + 255) / 256, 256, 0, stream>>>(pos, val, Aws);
    bias_kernel<<<OC_, 32, 0, stream>>>(val, bias);
    pwconv_wmma_kernel<<<NTOT / TILE_N, 128, 0, stream>>>(x, pos, Aws, bias, out);
}
